// GPT_30700426232401
// MI455X (gfx1250) — matrix operand order repaired and checked by reading, not compiled, not run on hardware
//
#include <hip/hip_runtime.h>
#include <hip/hip_bf16.h>

// ---------------- model dims (hardcoded from reference) ----------------
#define Dm   768
#define DFFm 3072
#define Tm   1024
#define Bm   2
#define Vm   32000
#define Lm   12
#define Hm   12
#define HDm  64
#define Mm   (Bm * Tm)   // 2048 token rows

typedef __attribute__((ext_vector_type(16))) _Float16 v16h;
typedef __attribute__((ext_vector_type(8)))  _Float16 v8h;
typedef __attribute__((ext_vector_type(8)))  float    v8f;

// epilogue flags
#define F_OUTF16 1
#define F_GELU   2
#define F_VTRANS 4

__device__ __forceinline__ v8f v8f_zero() {
  v8f z = {0.f, 0.f, 0.f, 0.f, 0.f, 0.f, 0.f, 0.f};
  return z;
}

__device__ __forceinline__ v16h concat8(v8h lo, v8h hi) {
  v16h r;
#pragma unroll
  for (int i = 0; i < 8; ++i) { r[i] = lo[i]; r[i + 8] = hi[i]; }
  return r;
}

// A-fragment (16x32 f16, MxK): lane holds row (lane&15).
// lanes 0-15: K = k0+0..15; lanes 16-31: K = k0+16..31.
__device__ __forceinline__ v16h load_a_frag(const _Float16* rowk0, int hi) {
  const _Float16* p = rowk0 + hi * 16;
  return concat8(*(const v8h*)p, *(const v8h*)(p + 8));
}

// B-fragment (32x16 f16, KxN): lane holds column (lane&15);
// lanes 0-15: K=k0+0..15, lanes 16-31: K=k0+16..31 (16 contiguous halves).
__device__ __forceinline__ v16h load_b_frag16(const _Float16* colk) {
  const v8h* p = (const v8h*)colk;
  return concat8(p[0], p[1]);
}

__device__ __forceinline__ v8f wmma_f16(v16h a, v16h b, v8f c) {
  return __builtin_amdgcn_wmma_f32_16x16x32_f16(false, a, false, b, (short)0, c,
                                                false, false);
}

// ---------------- embedding: x = tok_emb[idx] + pos_emb ----------------
__global__ __launch_bounds__(256) void embed_kernel(
    const int* __restrict__ idx, const float* __restrict__ tok,
    const float* __restrict__ pos, float* __restrict__ x) {
  int row = blockIdx.x;            // 0..Mm-1
  int t = row & (Tm - 1);
  int token = idx[row];
  const float* tr = tok + (size_t)token * Dm;
  const float* pr = pos + (size_t)t * Dm;
  float* xr = x + (size_t)row * Dm;
  for (int i = threadIdx.x; i < Dm; i += 256) xr[i] = tr[i] + pr[i];
}

// ---------------- layernorm (fp32 in -> f16 out) ----------------
__global__ __launch_bounds__(256) void ln_kernel(
    const float* __restrict__ x, const float* __restrict__ w,
    const float* __restrict__ bb, _Float16* __restrict__ out) {
  int row = blockIdx.x;
  const float* xr = x + (size_t)row * Dm;
  int t = threadIdx.x;
  float v0 = xr[t], v1 = xr[t + 256], v2 = xr[t + 512];
  __shared__ float sh[256];
  __shared__ float sh2[256];
  sh[t] = v0 + v1 + v2;
  sh2[t] = v0 * v0 + v1 * v1 + v2 * v2;
  __syncthreads();
  for (int off = 128; off > 0; off >>= 1) {
    if (t < off) { sh[t] += sh[t + off]; sh2[t] += sh2[t + off]; }
    __syncthreads();
  }
  float mean = sh[0] * (1.f / Dm);
  float var = sh2[0] * (1.f / Dm) - mean * mean;
  float rstd = rsqrtf(var + 1e-5f);
  _Float16* orow = out + (size_t)row * Dm;
  orow[t]       = (_Float16)((v0 - mean) * rstd * w[t]       + bb[t]);
  orow[t + 256] = (_Float16)((v1 - mean) * rstd * w[t + 256] + bb[t + 256]);
  orow[t + 512] = (_Float16)((v2 - mean) * rstd * w[t + 512] + bb[t + 512]);
}

// ------- weight transpose+downconvert: W[K,N] fp32 -> Wt[N,K] f16 -------
__global__ __launch_bounds__(256) void transposeW_kernel(
    const float* __restrict__ W, _Float16* __restrict__ Wt, int K, int N) {
  __shared__ float tile[32][33];
  int nb = blockIdx.x * 32, kb = blockIdx.y * 32;
  int tx = threadIdx.x, ty = threadIdx.y;  // 32 x 8
#pragma unroll
  for (int i = 0; i < 32; i += 8)
    tile[ty + i][tx] = W[(size_t)(kb + ty + i) * N + nb + tx];
  __syncthreads();
#pragma unroll
  for (int i = 0; i < 32; i += 8)
    Wt[(size_t)(nb + ty + i) * K + kb + tx] = (_Float16)tile[tx][ty + i];
}

// ---------------- WMMA GEMM: out = A[M,K] * Bt[N,K]^T + bias ----------------
// 4 waves / block; each wave -> 32 rows x 64 cols (2 A-frags x 4 B-frags).
__global__ __launch_bounds__(128) void wmma_gemm_kernel(
    const _Float16* __restrict__ A, const _Float16* __restrict__ Bt,
    const float* __restrict__ bias, const float* __restrict__ resid,
    void* __restrict__ out, int M, int N, int K, int flags) {
  int wave = threadIdx.x >> 5;
  int lane = threadIdx.x & 31;
  int hi = lane >> 4, ln = lane & 15;
  int mrow = blockIdx.y * 32;
  int nbase = blockIdx.x * 256 + wave * 64;

  const _Float16* ar0 = A + (size_t)(mrow + ln) * K;
  const _Float16* ar1 = A + (size_t)(mrow + 16 + ln) * K;

  v8f acc[2][4];
#pragma unroll
  for (int i = 0; i < 2; ++i)
#pragma unroll
    for (int n = 0; n < 4; ++n) acc[i][n] = v8f_zero();

  for (int k0 = 0; k0 < K; k0 += 32) {
    v16h a0 = load_a_frag(ar0 + k0, hi);
    v16h a1 = load_a_frag(ar1 + k0, hi);
#pragma unroll
    for (int n = 0; n < 4; ++n) {
      const _Float16* br = Bt + (size_t)(nbase + n * 16 + ln) * K + k0 + hi * 16;
      v16h bf = load_b_frag16(br);
      acc[0][n] = wmma_f16(a0, bf, acc[0][n]);
      acc[1][n] = wmma_f16(a1, bf, acc[1][n]);
    }
  }

#pragma unroll
  for (int i = 0; i < 2; ++i)
#pragma unroll
    for (int n = 0; n < 4; ++n)
#pragma unroll
      for (int r = 0; r < 8; ++r) {
        int m = mrow + i * 16 + r + 8 * hi;
        int nn = nbase + n * 16 + ln;
        float val = acc[i][n][r];
        if (bias) val += bias[nn];
        if (flags & F_GELU)
          val = 0.5f * val * (1.0f + erff(val * 0.70710678118654752f));
        if (flags & F_VTRANS) {
          // out = vT[B*H][64][T] f16 ; m = b*T + t ; nn = h*64 + d
          int b = m >> 10, t = m & (Tm - 1);
          int hh = nn >> 6, d = nn & 63;
          ((_Float16*)out)[(((size_t)(b * Hm + hh)) * HDm + d) * Tm + t] =
              (_Float16)val;
        } else if (flags & F_OUTF16) {
          ((_Float16*)out)[(size_t)m * N + nn] = (_Float16)val;
        } else {
          float* op = (float*)out;
          float rv = resid ? resid[(size_t)m * N + nn] : 0.f;
          op[(size_t)m * N + nn] = val + rv;
        }
      }
}

// ---------------- flash attention: 1 wave per (b,h,16 query rows) ----------
__global__ __launch_bounds__(32) void attn_kernel(
    const _Float16* __restrict__ q, const _Float16* __restrict__ k,
    const _Float16* __restrict__ vT, _Float16* __restrict__ y) {
  int blk = blockIdx.x;
  int qb = blk & 63;        // T/16 = 64 query blocks
  int bh = blk >> 6;
  int b = bh / Hm, h = bh % Hm;
  int lane = threadIdx.x;
  int hi = lane >> 4, ln = lane & 15;
  int q0 = qb * 16;

  const _Float16* qrow = q + ((size_t)(b * Tm + q0 + ln)) * Dm + h * HDm;
  v16h aq0 = load_a_frag(qrow, hi);
  v16h aq1 = load_a_frag(qrow + 32, hi);

  float mrow[8], lrow[8];
  v8f o[4];
#pragma unroll
  for (int r = 0; r < 8; ++r) { mrow[r] = -3.0e38f; lrow[r] = 0.f; }
#pragma unroll
  for (int n = 0; n < 4; ++n) o[n] = v8f_zero();

  __shared__ _Float16 lp[16][32];

  int nsteps = (q0 + 15) / 32 + 1;
  for (int s = 0; s < nsteps; ++s) {
    int kbb = s * 32;
    // ---- scores S = Q K^T for 32 keys (two 16-wide tiles) ----
    const _Float16* kr0 = k + ((size_t)(b * Tm + kbb + ln)) * Dm + h * HDm + hi * 16;
    const _Float16* kr1 = kr0 + (size_t)16 * Dm;
    v16h b00 = load_b_frag16(kr0);
    v16h b01 = load_b_frag16(kr0 + 32);
    v16h b10 = load_b_frag16(kr1);
    v16h b11 = load_b_frag16(kr1 + 32);
    v8f c0 = v8f_zero(), c1 = v8f_zero();
    c0 = wmma_f16(aq0, b00, c0);
    c0 = wmma_f16(aq1, b01, c0);
    c1 = wmma_f16(aq0, b10, c1);
    c1 = wmma_f16(aq1, b11, c1);

    // ---- causal mask + online softmax (per row r of this lane's group) ----
#pragma unroll
    for (int r = 0; r < 8; ++r) {
      int row = q0 + 8 * hi + r;
      float s0 = c0[r] * 0.125f;                 // 1/sqrt(64)
      float s1 = c1[r] * 0.125f;
      if (kbb + ln > row)      s0 = -3.0e38f;
      if (kbb + 16 + ln > row) s1 = -3.0e38f;
      float tm = fmaxf(s0, s1);
      tm = fmaxf(tm, __shfl_xor(tm, 1));
      tm = fmaxf(tm, __shfl_xor(tm, 2));
      tm = fmaxf(tm, __shfl_xor(tm, 4));
      tm = fmaxf(tm, __shfl_xor(tm, 8));
      float nm = fmaxf(mrow[r], tm);
      float alpha = __expf(mrow[r] - nm);
      float p0 = __expf(s0 - nm);
      float p1 = __expf(s1 - nm);
      float rs = p0 + p1;
      rs += __shfl_xor(rs, 1);
      rs += __shfl_xor(rs, 2);
      rs += __shfl_xor(rs, 4);
      rs += __shfl_xor(rs, 8);
      lrow[r] = lrow[r] * alpha + rs;
      mrow[r] = nm;
#pragma unroll
      for (int n = 0; n < 4; ++n) o[n][r] *= alpha;
      lp[8 * hi + r][ln] = (_Float16)p0;
      lp[8 * hi + r][16 + ln] = (_Float16)p1;
    }
    __syncthreads();  // single-wave block: LDS RAW ordering

    // ---- P (16x32 f16) as A-fragment, accumulate O += P * V ----
    const _Float16* lr = &lp[ln][hi * 16];
    v16h ap = concat8(*(const v8h*)lr, *(const v8h*)(lr + 8));
#pragma unroll
    for (int n = 0; n < 4; ++n) {
      const _Float16* vr =
          vT + ((size_t)bh * HDm + n * 16 + ln) * Tm + kbb + hi * 16;
      v16h bv = load_b_frag16(vr);
      o[n] = wmma_f16(ap, bv, o[n]);
    }
    __syncthreads();  // protect lp before next iteration
  }

  // ---- epilogue: y[b,t,h*64+d] = O / l ----
#pragma unroll
  for (int n = 0; n < 4; ++n)
#pragma unroll
    for (int r = 0; r < 8; ++r) {
      int row = q0 + 8 * hi + r;
      float val = o[n][r] / lrow[r];
      y[((size_t)(b * Tm + row)) * Dm + h * HDm + n * 16 + ln] = (_Float16)val;
    }
}

// ---------------- host orchestration ----------------
static void launch_transpose(const float* W, _Float16* Wt, int K, int N,
                             hipStream_t s) {
  dim3 g(N / 32, K / 32), blk(32, 8);
  transposeW_kernel<<<g, blk, 0, s>>>(W, Wt, K, N);
}

static void launch_gemm(const _Float16* A, const _Float16* Bt, const float* bias,
                        const float* resid, void* out, int M, int N, int K,
                        int flags, hipStream_t s) {
  dim3 g(N / 256, M / 32);
  wmma_gemm_kernel<<<g, dim3(128), 0, s>>>(A, Bt, bias, resid, out, M, N, K,
                                           flags);
}

extern "C" void kernel_launch(void* const* d_in, const int* in_sizes, int n_in,
                              void* d_out, int out_size, void* d_ws,
                              size_t ws_size, hipStream_t stream) {
  (void)in_sizes; (void)n_in; (void)out_size; (void)ws_size;

  const int*   idx     = (const int*)d_in[0];
  const float* tok_emb = (const float*)d_in[1];
  const float* pos_emb = (const float*)d_in[2];
  const float* ln1_w   = (const float*)d_in[3];
  const float* ln1_b   = (const float*)d_in[4];
  const float* Wq      = (const float*)d_in[5];
  const float* bq      = (const float*)d_in[6];
  const float* Wk      = (const float*)d_in[7];
  const float* bk      = (const float*)d_in[8];
  const float* Wv      = (const float*)d_in[9];
  const float* bv      = (const float*)d_in[10];
  const float* Wo      = (const float*)d_in[11];
  const float* bo      = (const float*)d_in[12];
  const float* ln2_w   = (const float*)d_in[13];
  const float* ln2_b   = (const float*)d_in[14];
  const float* W1      = (const float*)d_in[15];
  const float* b1      = (const float*)d_in[16];
  const float* W2      = (const float*)d_in[17];
  const float* b2      = (const float*)d_in[18];
  const float* lnf_w   = (const float*)d_in[19];
  const float* lnf_b   = (const float*)d_in[20];
  const float* head_w  = (const float*)d_in[21];

  // workspace layout (all sizes multiple of 256B)
  char* p = (char*)d_ws;
  float*    x    = (float*)p;    p += (size_t)Mm * Dm * 4;
  _Float16* h16  = (_Float16*)p; p += (size_t)Mm * Dm * 2;
  _Float16* q16  = (_Float16*)p; p += (size_t)Mm * Dm * 2;
  _Float16* k16  = (_Float16*)p; p += (size_t)Mm * Dm * 2;
  _Float16* vT16 = (_Float16*)p; p += (size_t)Mm * Dm * 2;
  _Float16* y16  = (_Float16*)p; p += (size_t)Mm * Dm * 2;
  _Float16* m16  = (_Float16*)p; p += (size_t)Mm * DFFm * 2;
  _Float16* wT   = (_Float16*)p; p += (size_t)Vm * Dm * 2;

  embed_kernel<<<Mm, 256, 0, stream>>>(idx, tok_emb, pos_emb, x);

  for (int l = 0; l < Lm; ++l) {
    const float* wq = Wq + (size_t)l * Dm * Dm;
    const float* wk = Wk + (size_t)l * Dm * Dm;
    const float* wv = Wv + (size_t)l * Dm * Dm;
    const float* wo = Wo + (size_t)l * Dm * Dm;
    const float* w1 = W1 + (size_t)l * Dm * DFFm;
    const float* w2 = W2 + (size_t)l * DFFm * Dm;

    // h = LN1(x)
    ln_kernel<<<Mm, 256, 0, stream>>>(x, ln1_w + l * Dm, ln1_b + l * Dm, h16);

    // q,k,v projections (v written pre-transposed per head: vT[B*H,64,T])
    launch_transpose(wq, wT, Dm, Dm, stream);
    launch_gemm(h16, wT, bq + l * Dm, nullptr, q16, Mm, Dm, Dm, F_OUTF16, stream);
    launch_transpose(wk, wT, Dm, Dm, stream);
    launch_gemm(h16, wT, bk + l * Dm, nullptr, k16, Mm, Dm, Dm, F_OUTF16, stream);
    launch_transpose(wv, wT, Dm, Dm, stream);
    launch_gemm(h16, wT, bv + l * Dm, nullptr, vT16, Mm, Dm, Dm, F_VTRANS, stream);

    // flash attention -> y16
    attn_kernel<<<Bm * Hm * (Tm / 16), 32, 0, stream>>>(q16, k16, vT16, y16);

    // x = x + y @ Wo + bo
    launch_transpose(wo, wT, Dm, Dm, stream);
    launch_gemm(y16, wT, bo + l * Dm, x, x, Mm, Dm, Dm, 0, stream);

    // h = LN2(x); mid = gelu(h @ W1 + b1); x = x + mid @ W2 + b2
    ln_kernel<<<Mm, 256, 0, stream>>>(x, ln2_w + l * Dm, ln2_b + l * Dm, h16);
    launch_transpose(w1, wT, Dm, DFFm, stream);
    launch_gemm(h16, wT, b1 + l * DFFm, nullptr, m16, Mm, DFFm, Dm,
                F_OUTF16 | F_GELU, stream);
    launch_transpose(w2, wT, DFFm, Dm, stream);
    launch_gemm(m16, wT, b2 + l * Dm, x, x, Mm, Dm, DFFm, 0, stream);
  }

  // final LN + head: logits = LN(x) @ head_w   (fp32 out, no bias)
  ln_kernel<<<Mm, 256, 0, stream>>>(x, lnf_w, lnf_b, h16);
  launch_transpose(head_w, wT, Dm, Vm, stream);
  launch_gemm(h16, wT, nullptr, nullptr, (float*)d_out, Mm, Vm, Dm, 0, stream);
}
